// HybridAttention_60026462929522
// MI455X (gfx1250) — matrix operand order repaired and checked by reading, not compiled, not run on hardware
//
#include <hip/hip_runtime.h>

typedef unsigned short u16;
typedef unsigned int u32;

typedef __bf16 bf16x16 __attribute__((ext_vector_type(16)));
typedef float f32x8 __attribute__((ext_vector_type(8)));

#define Bb 16
#define Lk 1024
#define HIDk 1024
#define NFREQ 308
#define NFPAD 320
#define FLEFT 205
#define TOPK 6

__device__ __forceinline__ u16 f2bf(float f) {
  u32 u = __float_as_uint(f);
  u32 r = (u + 0x7FFFu + ((u >> 16) & 1u)) >> 16;
  return (u16)r;
}
__device__ __forceinline__ float bf2f(u16 h) {
  return __uint_as_float(((u32)h) << 16);
}

// Wave-relative LDS byte offset (what GLOBAL_LOAD_ASYNC_TO_LDS's VDST expects).
__device__ __forceinline__ u32 lds_off(const void* p) {
  return (u32)(size_t)(__attribute__((address_space(3))) const void*)p;
}

// ---------------------------------------------------------------- conversions
__global__ void cvt_f32_to_bf16(const float* __restrict__ src, u16* __restrict__ dst, int n) {
  for (int i = blockIdx.x * blockDim.x + threadIdx.x; i < n; i += gridDim.x * blockDim.x)
    dst[i] = f2bf(src[i]);
}

// Band DFT twiddles, freq-major A matrix (NFPAD x L), rows >= NFREQ zero.
// Twr = cos(theta), Twi = -sin(theta) so QFr = Re(rfft), QFi = Im(rfft).
__global__ void build_twiddles_kernel(u16* __restrict__ Twr, u16* __restrict__ Twi) {
  int f = blockIdx.x;   // 0..NFPAD-1
  int l = threadIdx.x;  // 0..L-1
  float cr = 0.f, ci = 0.f;
  if (f < NFREQ) {
    int freq = FLEFT + f;
    int ph = (freq * l) & (Lk - 1);  // exact phase reduction mod L
    float ang = -6.28318530717958647692f * (float)ph * (1.0f / (float)Lk);
    float s, c;
    __sincosf(ang, &s, &c);
    cr = c; ci = s;  // s == -sin(theta)
  }
  Twr[(size_t)f * Lk + l] = f2bf(cr);
  Twi[(size_t)f * Lk + l] = f2bf(ci);
}

// ---------------------------------------------------------------- WMMA GEMM
// C[m,n] = sum_k A[m,k]*B[k,n]   A: bf16 row-major MxK, B: bf16 row-major KxN
// MODE 0: out bf16  = acc + bias[n]
// MODE 1: out f32   = acc
// MODE 2: out f32   = acc + bias[n] + resid[m*N+n]
#define BM 128
#define BN 128
#define BKK 32
#define SA_STRIDE 40
#define SB_STRIDE 136

union Frag16 { bf16x16 v; u16 u[16]; uint4 q[2]; };
union Acc8 { f32x8 v; float f[8]; };

template <int MODE>
__global__ __launch_bounds__(256) void wmma_gemm_bf16(
    const u16* __restrict__ A, size_t strideAz,
    const u16* __restrict__ Bm, size_t strideBz,
    const float* __restrict__ bias,
    const float* __restrict__ resid,
    void* __restrict__ Cout, size_t strideCz,
    int M, int N, int K) {
  __shared__ u16 sA[2][BM * SA_STRIDE];
  __shared__ u16 sB[2][BKK * SB_STRIDE];

  const int z = blockIdx.z;
  A += (size_t)z * strideAz;
  Bm += (size_t)z * strideBz;

  const int tid = threadIdx.x;
  const int lane = tid & 31;
  const int wave = tid >> 5;        // 8 waves: 2 (M) x 4 (N)
  const int waveM = wave >> 2;      // 0..1 -> 64 rows each
  const int waveN = wave & 3;       // 0..3 -> 32 cols each
  const int m_block = blockIdx.y * BM;
  const int n_block = blockIdx.x * BN;

  // staging mapping: each thread owns one contiguous 32B chunk of A and of B
  const int ast_r = tid >> 1;   // 0..127 row of A tile
  const int ast_h = tid & 1;    // which 16-elem half of the 32-wide A tile
  const int bst_k = tid >> 3;   // 0..31 row of B tile
  const int bst_p = tid & 7;    // 16-elem col group of the 128-wide B tile

  // Clamp OOB A rows instead of skipping: keeps the async issue count uniform
  // per wave (so s_wait_asynccnt thresholds are exact); C rows >= M are never
  // stored, so the garbage they accumulate is harmless.
  int aRow = m_block + ast_r;
  if (aRow >= M) aRow = M - 1;
  const u16* gA = A + (size_t)aRow * K + ast_h * 16;
  const u16* gB = Bm + (size_t)bst_k * N + n_block + bst_p * 16;

  // Issue one tile's staging as 4 async global->LDS copies (ASYNCcnt += 4).
  // INST_OFFSET is added to BOTH the LDS and global addresses per the ISA, so
  // offset:16 moves the second 16B sub-chunk of each 32B chunk.
  auto issue_tile = [&](int k0, int buf) {
    const u16* pa = gA + k0;
    u32 la = lds_off(&sA[buf][ast_r * SA_STRIDE + ast_h * 16]);
    asm volatile("global_load_async_to_lds_b128 %0, %1, off"
                 :: "v"(la), "v"(pa) : "memory");
    asm volatile("global_load_async_to_lds_b128 %0, %1, off offset:16"
                 :: "v"(la), "v"(pa) : "memory");
    const u16* pb = gB + (size_t)k0 * N;
    u32 lb = lds_off(&sB[buf][bst_k * SB_STRIDE + bst_p * 16]);
    asm volatile("global_load_async_to_lds_b128 %0, %1, off"
                 :: "v"(lb), "v"(pb) : "memory");
    asm volatile("global_load_async_to_lds_b128 %0, %1, off offset:16"
                 :: "v"(lb), "v"(pb) : "memory");
  };

  Acc8 acc[4][2];
#pragma unroll
  for (int mt = 0; mt < 4; ++mt)
#pragma unroll
    for (int nt = 0; nt < 2; ++nt)
#pragma unroll
      for (int r = 0; r < 8; ++r) acc[mt][nt].f[r] = 0.f;

  const int arow = lane & 15;
  const int khalf = lane >> 4;  // 0/1 selects K sub-block per ISA A layout

  issue_tile(0, 0);
  const int iters = K / BKK;
  for (int it = 0; it < iters; ++it) {
    const int buf = it & 1;
    if (it + 1 < iters) {
      issue_tile((it + 1) * BKK, buf ^ 1);
      if (it + 2 < iters) {  // pull tile i+2 toward L2 (global_prefetch_b8)
        __builtin_prefetch(gA + (it + 2) * BKK, 0, 1);
        __builtin_prefetch(gB + (size_t)(it + 2) * BKK * N, 0, 1);
      }
      // Async loads complete in order: <=4 outstanding => tile `it` resident.
      asm volatile("s_wait_asynccnt 0x4" ::: "memory");
    } else {
      asm volatile("s_wait_asynccnt 0x0" ::: "memory");
    }
    __syncthreads();  // tile `it` resident for every wave in the group

    // ---- fragments per ISA VGPR layouts ----
    Frag16 afr[4], bfr[2];
#pragma unroll
    for (int mt = 0; mt < 4; ++mt) {
      const u16* p = &sA[buf][(waveM * 64 + mt * 16 + arow) * SA_STRIDE + khalf * 8];
      afr[mt].q[0] = *(const uint4*)p;          // K = khalf*8 + 0..7
      afr[mt].q[1] = *(const uint4*)(p + 16);   // K = 16 + khalf*8 + 0..7
    }
#pragma unroll
    for (int nt = 0; nt < 2; ++nt) {
      const u16* p = &sB[buf][(lane >> 4) * 8 * SB_STRIDE + waveN * 32 + nt * 16 + (lane & 15)];
#pragma unroll
      for (int s = 0; s < 8; ++s) {
        bfr[nt].u[s] = p[s * SB_STRIDE];              // K = khalf*8 + s (column N = lane&15)
        bfr[nt].u[8 + s] = p[(16 + s) * SB_STRIDE];   // K = 16 + khalf*8 + s
      }
    }
#pragma unroll
    for (int mt = 0; mt < 4; ++mt)
#pragma unroll
      for (int nt = 0; nt < 2; ++nt)
        acc[mt][nt].v = __builtin_amdgcn_wmma_f32_16x16x32_bf16(
            false, afr[mt].v, false, bfr[nt].v, (short)0, acc[mt][nt].v, false, false);
    __syncthreads();  // everyone done reading buf before it is refilled
  }

  // ---- epilogue: C 16x16 f32 layout (VGPR r = M row, lane-half adds 8) ----
  const int mhi = (lane >> 4) * 8;
  const int ncol = n_block + waveN * 32 + (lane & 15);
#pragma unroll
  for (int mt = 0; mt < 4; ++mt) {
#pragma unroll
    for (int nt = 0; nt < 2; ++nt) {
      const int colw = ncol + nt * 16;
#pragma unroll
      for (int r = 0; r < 8; ++r) {
        const int grow = m_block + waveM * 64 + mt * 16 + mhi + r;
        if (grow < M) {
          float val = acc[mt][nt].f[r];
          if (MODE == 0) {
            ((u16*)Cout)[(size_t)z * strideCz + (size_t)grow * N + colw] = f2bf(val + bias[colw]);
          } else if (MODE == 1) {
            ((float*)Cout)[(size_t)z * strideCz + (size_t)grow * N + colw] = val;
          } else {
            ((float*)Cout)[(size_t)grow * N + colw] =
                val + bias[colw] + resid[(size_t)grow * N + colw];
          }
        }
      }
    }
  }
}

// ------------------------------------------------- channel-sum band product
// S[b,f] = sum_c qf[b,f,c] * conj(kf[b,f,c])
__global__ __launch_bounds__(256) void band_reduce_kernel(
    const float* __restrict__ QFr, const float* __restrict__ QFi,
    const float* __restrict__ KFr, const float* __restrict__ KFi,
    float* __restrict__ Sr, float* __restrict__ Si) {
  const int f = blockIdx.x;  // 0..NFREQ-1
  const int b = blockIdx.y;
  const size_t base = ((size_t)b * NFPAD + f) * HIDk;
  float ar = 0.f, ai = 0.f;
  for (int c = threadIdx.x; c < HIDk; c += 256) {
    float qr = QFr[base + c], qi = QFi[base + c];
    float kr = KFr[base + c], ki = KFi[base + c];
    ar += qr * kr + qi * ki;
    ai += qi * kr - qr * ki;
  }
  __shared__ float sr_[256], si_[256];
  int t = threadIdx.x;
  sr_[t] = ar; si_[t] = ai;
  __syncthreads();
  for (int s = 128; s > 0; s >>= 1) {
    if (t < s) { sr_[t] += sr_[t + s]; si_[t] += si_[t + s]; }
    __syncthreads();
  }
  if (t == 0) {
    Sr[b * NFPAD + f] = sr_[0];
    Si[b * NFPAD + f] = si_[0];
  }
}

// ------------------------------------------------- inverse band transform
// mean_value[b,l] = (1/(L*HID)) * [ 2*sum_{f<512}(Sr cos - Si sin) + Nyquist ]
__global__ __launch_bounds__(256) void mean_value_kernel(
    const float* __restrict__ Sr, const float* __restrict__ Si,
    float* __restrict__ meanv) {
  const int b = blockIdx.x;
  __shared__ float shr[NFREQ], shi[NFREQ];
  for (int i = threadIdx.x; i < NFREQ; i += 256) {
    shr[i] = Sr[b * NFPAD + i];
    shi[i] = Si[b * NFPAD + i];
  }
  __syncthreads();
  for (int l = threadIdx.x; l < Lk; l += 256) {
    float accv = 0.f;
    for (int f = 0; f < NFREQ; ++f) {
      int freq = FLEFT + f;
      int ph = (freq * l) & (Lk - 1);
      float ang = 6.28318530717958647692f * (float)ph * (1.0f / (float)Lk);
      float s, c;
      __sincosf(ang, &s, &c);
      float w = (freq == Lk / 2) ? 1.f : 2.f;
      accv += w * (shr[f] * c - shi[f] * s);
    }
    meanv[(size_t)b * Lk + l] = accv * (1.0f / ((float)Lk * (float)HIDk));
  }
}

// ------------------------------------------------- top-k + per-batch softmax
__global__ __launch_bounds__(1024) void topk_softmax_kernel(
    const float* __restrict__ meanv, int* __restrict__ idxOut, float* __restrict__ wOut) {
  const int t = threadIdx.x;
  __shared__ float g[Lk];
  __shared__ float v[Lk];
  __shared__ int ix[Lk];
  __shared__ int sidx[8];
  float s = 0.f;
  for (int b = 0; b < Bb; ++b) s += meanv[(size_t)b * Lk + t];
  g[t] = s * (1.0f / (float)Bb);
  __syncthreads();
  for (int r = 0; r < TOPK; ++r) {
    v[t] = g[t]; ix[t] = t;
    __syncthreads();
    for (int st = 512; st > 0; st >>= 1) {
      if (t < st) {
        if (v[t + st] > v[t] || (v[t + st] == v[t] && ix[t + st] < ix[t])) {
          v[t] = v[t + st];
          ix[t] = ix[t + st];
        }
      }
      __syncthreads();
    }
    if (t == 0) {
      sidx[r] = ix[0];
      g[ix[0]] = -3.402823466e38f;
    }
    __syncthreads();
  }
  if (t < Bb) {
    float e[TOPK];
    float mx = -3.402823466e38f;
    for (int i = 0; i < TOPK; ++i) {
      e[i] = meanv[(size_t)t * Lk + sidx[i]];
      mx = fmaxf(mx, e[i]);
    }
    float sum = 0.f;
    for (int i = 0; i < TOPK; ++i) { e[i] = __expf(e[i] - mx); sum += e[i]; }
    for (int i = 0; i < TOPK; ++i) wOut[t * 8 + i] = e[i] / sum;
  }
  if (t < TOPK) idxOut[t] = sidx[t];
}

// ------------------------------------------------- delay aggregation
// ctx[b,l,c] = sum_i w[b,i] * v[b, (l+idx[i]) mod L, c]
__global__ __launch_bounds__(256) void delay_agg_kernel(
    const u16* __restrict__ vbf, const int* __restrict__ idx,
    const float* __restrict__ w, u16* __restrict__ ctx) {
  const int bl = blockIdx.x;
  const int b = bl >> 10;
  const int l = bl & (Lk - 1);
  __shared__ float sw[TOPK];
  __shared__ int sid[TOPK];
  if (threadIdx.x < TOPK) {
    sw[threadIdx.x] = w[b * 8 + threadIdx.x];
    sid[threadIdx.x] = idx[threadIdx.x];
  }
  __syncthreads();
  const size_t bbase = (size_t)b * Lk * HIDk;
  const int c = threadIdx.x * 4;
  float a0 = 0.f, a1 = 0.f, a2 = 0.f, a3 = 0.f;
#pragma unroll
  for (int i = 0; i < TOPK; ++i) {
    const int ls = (l + sid[i]) & (Lk - 1);
    const u16* vr = vbf + bbase + (size_t)ls * HIDk + c;
    const float wi = sw[i];
    a0 += wi * bf2f(vr[0]);
    a1 += wi * bf2f(vr[1]);
    a2 += wi * bf2f(vr[2]);
    a3 += wi * bf2f(vr[3]);
  }
  u16* o = ctx + bbase + (size_t)l * HIDk + c;
  o[0] = f2bf(a0); o[1] = f2bf(a1); o[2] = f2bf(a2); o[3] = f2bf(a3);
}

// ------------------------------------------------- layernorm
__global__ __launch_bounds__(256) void layernorm_kernel(
    const float* __restrict__ X, const float* __restrict__ gamma,
    const float* __restrict__ beta, float* __restrict__ out) {
  const size_t row = blockIdx.x;
  const float* x = X + row * HIDk;
  float vals[4];
  float lsum = 0.f, lsq = 0.f;
#pragma unroll
  for (int j = 0; j < 4; ++j) {
    float vv = x[threadIdx.x + 256 * j];
    vals[j] = vv;
    lsum += vv;
    lsq += vv * vv;
  }
  __shared__ float ssum[256], ssq[256];
  int t = threadIdx.x;
  ssum[t] = lsum; ssq[t] = lsq;
  __syncthreads();
  for (int s = 128; s > 0; s >>= 1) {
    if (t < s) { ssum[t] += ssum[t + s]; ssq[t] += ssq[t + s]; }
    __syncthreads();
  }
  const float mean = ssum[0] * (1.0f / (float)HIDk);
  const float var = ssq[0] * (1.0f / (float)HIDk) - mean * mean;
  const float inv = rsqrtf(var + 1e-12f);
#pragma unroll
  for (int j = 0; j < 4; ++j) {
    int cc = threadIdx.x + 256 * j;
    out[row * HIDk + cc] = (vals[j] - mean) * inv * gamma[cc] + beta[cc];
  }
}

// ================================================================ host side
extern "C" void kernel_launch(void* const* d_in, const int* in_sizes, int n_in,
                              void* d_out, int out_size, void* d_ws, size_t ws_size,
                              hipStream_t stream) {
  (void)in_sizes; (void)n_in; (void)out_size; (void)ws_size;
  const float* X = (const float*)d_in[0];
  // d_in[1] = attention_mask: zeros, never used by the reference math
  const float* Wq = (const float*)d_in[2];
  const float* bq = (const float*)d_in[3];
  const float* Wk = (const float*)d_in[4];
  const float* bk = (const float*)d_in[5];
  const float* Wv = (const float*)d_in[6];
  const float* bv = (const float*)d_in[7];
  const float* Wd = (const float*)d_in[8];
  const float* bd = (const float*)d_in[9];
  const float* gamma = (const float*)d_in[10];
  const float* beta = (const float*)d_in[11];
  float* out = (float*)d_out;

  const size_t BLH = (size_t)Bb * Lk * HIDk;  // 16M elements
  const size_t WW = (size_t)HIDk * HIDk;      // 1M elements

  char* ws = (char*)d_ws;
  size_t cur = 0;
  auto alloc = [&](size_t bytes) -> void* {
    void* p = ws + cur;
    cur = (cur + bytes + 255) & ~(size_t)255;
    return p;
  };

  u16* Xbf = (u16*)alloc(BLH * 2);
  u16* Wqb = (u16*)alloc(WW * 2);
  u16* Wkb = (u16*)alloc(WW * 2);
  u16* Wvb = (u16*)alloc(WW * 2);
  u16* Wdb = (u16*)alloc(WW * 2);
  u16* qbf = (u16*)alloc(BLH * 2);
  u16* kbf = (u16*)alloc(BLH * 2);
  u16* vbf = (u16*)alloc(BLH * 2);
  u16* ctxbf = (u16*)alloc(BLH * 2);
  u16* Twr = (u16*)alloc((size_t)NFPAD * Lk * 2);
  u16* Twi = (u16*)alloc((size_t)NFPAD * Lk * 2);
  float* QFr = (float*)alloc((size_t)Bb * NFPAD * HIDk * 4);
  float* QFi = (float*)alloc((size_t)Bb * NFPAD * HIDk * 4);
  float* KFr = (float*)alloc((size_t)Bb * NFPAD * HIDk * 4);
  float* KFi = (float*)alloc((size_t)Bb * NFPAD * HIDk * 4);
  float* Sr = (float*)alloc((size_t)Bb * NFPAD * 4);
  float* Si = (float*)alloc((size_t)Bb * NFPAD * 4);
  float* meanv = (float*)alloc((size_t)Bb * Lk * 4);
  int* idxbuf = (int*)alloc(64);
  float* corrw = (float*)alloc((size_t)Bb * 8 * 4);
  float* preln = (float*)alloc(BLH * 4);

  // 1) bf16 conversions + twiddle table
  cvt_f32_to_bf16<<<4096, 256, 0, stream>>>(X, Xbf, (int)BLH);
  cvt_f32_to_bf16<<<1024, 256, 0, stream>>>(Wq, Wqb, (int)WW);
  cvt_f32_to_bf16<<<1024, 256, 0, stream>>>(Wk, Wkb, (int)WW);
  cvt_f32_to_bf16<<<1024, 256, 0, stream>>>(Wv, Wvb, (int)WW);
  cvt_f32_to_bf16<<<1024, 256, 0, stream>>>(Wd, Wdb, (int)WW);
  build_twiddles_kernel<<<NFPAD, Lk, 0, stream>>>(Twr, Twi);

  // 2) Q/K/V projection GEMMs (WMMA bf16 -> bf16 with bias)
  dim3 bigGrid(HIDk / BN, (Bb * Lk) / BM, 1);
  wmma_gemm_bf16<0><<<bigGrid, 256, 0, stream>>>(Xbf, 0, Wqb, 0, bq, nullptr, qbf, 0,
                                                 Bb * Lk, HIDk, HIDk);
  wmma_gemm_bf16<0><<<bigGrid, 256, 0, stream>>>(Xbf, 0, Wkb, 0, bk, nullptr, kbf, 0,
                                                 Bb * Lk, HIDk, HIDk);
  wmma_gemm_bf16<0><<<bigGrid, 256, 0, stream>>>(Xbf, 0, Wvb, 0, bv, nullptr, vbf, 0,
                                                 Bb * Lk, HIDk, HIDk);

  // 3) band DFT as WMMA GEMM: QF[b][f][c] = Tw[f][l] @ q[b][l][c]
  dim3 dftGrid(HIDk / BN, (NFPAD + BM - 1) / BM, Bb);
  const size_t zB = (size_t)Lk * HIDk;
  const size_t zC = (size_t)NFPAD * HIDk;
  wmma_gemm_bf16<1><<<dftGrid, 256, 0, stream>>>(Twr, 0, qbf, zB, nullptr, nullptr, QFr, zC,
                                                 NFPAD, HIDk, Lk);
  wmma_gemm_bf16<1><<<dftGrid, 256, 0, stream>>>(Twi, 0, qbf, zB, nullptr, nullptr, QFi, zC,
                                                 NFPAD, HIDk, Lk);
  wmma_gemm_bf16<1><<<dftGrid, 256, 0, stream>>>(Twr, 0, kbf, zB, nullptr, nullptr, KFr, zC,
                                                 NFPAD, HIDk, Lk);
  wmma_gemm_bf16<1><<<dftGrid, 256, 0, stream>>>(Twi, 0, kbf, zB, nullptr, nullptr, KFi, zC,
                                                 NFPAD, HIDk, Lk);

  // 4) channel-summed band product, inverse transform, top-k + softmax
  band_reduce_kernel<<<dim3(NFREQ, Bb), 256, 0, stream>>>(QFr, QFi, KFr, KFi, Sr, Si);
  mean_value_kernel<<<Bb, 256, 0, stream>>>(Sr, Si, meanv);
  topk_softmax_kernel<<<1, 1024, 0, stream>>>(meanv, idxbuf, corrw);

  // 5) delay aggregation over v
  delay_agg_kernel<<<Bb * Lk, 256, 0, stream>>>(vbf, idxbuf, corrw, ctxbf);

  // 6) output projection fused with bias + residual, then layernorm
  wmma_gemm_bf16<2><<<bigGrid, 256, 0, stream>>>(ctxbf, 0, Wdb, 0, bd, X, preln, 0,
                                                 Bb * Lk, HIDk, HIDk);
  layernorm_kernel<<<Bb * Lk, 256, 0, stream>>>(preln, gamma, beta, out);
}
